// SigLIPLoss_2697239462690
// MI455X (gfx1250) — matrix operand order repaired and checked by reading, not compiled, not run on hardware
//
#include <hip/hip_runtime.h>

typedef __attribute__((ext_vector_type(16))) __bf16  v16bf;
typedef __attribute__((ext_vector_type(8)))  float   v8f;
typedef __attribute__((ext_vector_type(4)))  unsigned int u32x4;
typedef __attribute__((ext_vector_type(8)))  int     i32x8;
typedef __attribute__((ext_vector_type(4)))  int     i32x4;
typedef unsigned short u16;

#define NTOT  16384
#define DDIM  256
#define BM    128
#define BN    128
#define KC    32
#define NCHUNK (DDIM / KC)   // 8
#define PITCH 40             // 32 + 8 bf16 pad: 80B rows, 16B aligned, conflict-free stride

#define LOG2E 1.4426950408889634f
#define LN2   0.6931471805599453f

#if __has_builtin(__builtin_amdgcn_tensor_load_to_lds)
#define HAVE_TDM 1
#else
#define HAVE_TDM 0
#endif

union Frag {
    u32x4 u[2];
    v16bf v;
};

// ---------------- f32 -> bf16 (RNE) conversion ----------------
__device__ __forceinline__ u16 f2bf(float f) {
    unsigned int u = __float_as_uint(f);
    unsigned int r = (u + 0x7FFFu + ((u >> 16) & 1u)) >> 16;
    return (u16)r;
}

__global__ __launch_bounds__(256) void cvt_bf16(const float* __restrict__ a,
                                                const float* __restrict__ b,
                                                u16* __restrict__ oa,
                                                u16* __restrict__ ob) {
    size_t i = ((size_t)blockIdx.x * blockDim.x + threadIdx.x) * 8;
    if (i >= (size_t)NTOT * DDIM) return;
    #pragma unroll
    for (int s = 0; s < 2; ++s) {
        const float* src = s ? b : a;
        u16* dst = s ? ob : oa;
        u16 tmp[8];
        #pragma unroll
        for (int k = 0; k < 8; ++k) tmp[k] = f2bf(src[i + k]);
        *reinterpret_cast<u32x4*>(dst + i) = *reinterpret_cast<u32x4*>(tmp);
    }
}

// softplus in log2 units: max(s,0) + log2(1 + 2^-|s|); one v_exp_f32 + one v_log_f32
__device__ __forceinline__ float softplus_log2(float s) {
    float m = fmaxf(s, 0.f);
    float e = __builtin_amdgcn_exp2f(-fabsf(s));
    float l = __builtin_amdgcn_logf(1.0f + e);
    return m + l;
}

#if HAVE_TDM
// TDM: load a 128-row x 32-col bf16 tile (row stride 256 elems) into LDS,
// padding 16B after every 64B row -> LDS pitch = 80B = PITCH u16.
__device__ __forceinline__ void tdm_load_tile(const u16* gptr, unsigned ldsOff) {
    unsigned long long ga = (unsigned long long)(uintptr_t)gptr;
    u32x4 g0 = { 1u,                                   // count=1 valid descriptor
                 ldsOff,                               // lds_addr (bytes)
                 (unsigned)ga,                         // global_addr[31:0]
                 ((unsigned)(ga >> 32) & 0x01FFFFFFu) | 0x80000000u };  // ga[56:32] | type=2
    i32x8 g1 = { 0x06D10000,    // data_size=2B | pad_enable | pad_interval=3 | pad_amount=3
                 0x01000000,    // tensor_dim0 = 256 (bits 79:48 -> low16 in [31:16])
                 0x40000000,    // tensor_dim1 = 16384 (bits 111:80 -> low16 in [31:16])
                 0x00200000,    // tile_dim0 = 32 (bits 127:112)
                 128,           // tile_dim1 = 128, tile_dim2 = 0
                 256,           // tensor_dim0_stride = 256 (low 32)
                 0, 0 };
    i32x4 z4 = { 0, 0, 0, 0 };
#if defined(__clang_major__) && __clang_major__ >= 23
    i32x8 z8 = { 0, 0, 0, 0, 0, 0, 0, 0 };
    __builtin_amdgcn_tensor_load_to_lds(g0, g1, z4, z4, z8, 0);
#else
    __builtin_amdgcn_tensor_load_to_lds(g0, g1, z4, z4, 0);
#endif
}
#endif

// ---------------- fused GEMM + SigLIP loss partial ----------------
__global__ __launch_bounds__(256) void siglip_gemm_loss(const u16* __restrict__ Abf,
                                                        const u16* __restrict__ Bbf,
                                                        const float* __restrict__ biasPtr,
                                                        float* __restrict__ partials) {
    __shared__ u16 sA[2][BM * PITCH];
    __shared__ u16 sB[2][BN * PITCH];
    __shared__ float sred[8];

    const int t    = threadIdx.x;
    const int wid  = t >> 5;
    const int lane = t & 31;
    const int waveM = wid & 3;          // 4 waves along M
    const int waveN = wid >> 2;         // 2 waves along N
    const int wm = waveM * 32;          // wave tile: 32 (M) x 64 (N)
    const int wn = waveN * 64;
    const int half = lane >> 4;
    const int lr   = lane & 15;

    const u16* gA = Abf + (size_t)blockIdx.y * BM * DDIM;
    const u16* gB = Bbf + (size_t)blockIdx.x * BN * DDIM;

    const v8f vzero = {0.f, 0.f, 0.f, 0.f, 0.f, 0.f, 0.f, 0.f};
    v8f acc[2][4];
    #pragma unroll
    for (int tr = 0; tr < 2; ++tr)
        #pragma unroll
        for (int tc = 0; tc < 4; ++tc) acc[tr][tc] = vzero;

#if HAVE_TDM
    const unsigned offA0 = (unsigned)(uintptr_t)&sA[0][0];
    const unsigned offA1 = (unsigned)(uintptr_t)&sA[1][0];
    const unsigned offB0 = (unsigned)(uintptr_t)&sB[0][0];
    const unsigned offB1 = (unsigned)(uintptr_t)&sB[1][0];
    // prologue: chunk 0 -> buffer 0
    if (wid == 0) {
        tdm_load_tile(gA, offA0);
        tdm_load_tile(gB, offB0);
    }
#endif

    for (int kc = 0; kc < NCHUNK; ++kc) {
        const int cur = kc & 1;
#if HAVE_TDM
        __syncthreads();   // readers of buffer 'nxt' (iter kc-1) are done
        if (wid == 0) {
            if (kc + 1 < NCHUNK) {
                // issue next chunk into the other buffer, then wait for current
                tdm_load_tile(gA + (kc + 1) * KC, cur ? offA0 : offA1);
                tdm_load_tile(gB + (kc + 1) * KC, cur ? offB0 : offB1);
                __builtin_amdgcn_s_wait_tensorcnt(2);   // current chunk's pair done
            } else {
                __builtin_amdgcn_s_wait_tensorcnt(0);
            }
        }
        __syncthreads();
#else
        __syncthreads();
        for (int q = t; q < (BM * KC / 8); q += 256) {
            const int row = q / (KC / 8);
            const int c8  = (q % (KC / 8)) << 3;
            *reinterpret_cast<u32x4*>(&sA[cur][row * PITCH + c8]) =
                *reinterpret_cast<const u32x4*>(gA + (size_t)row * DDIM + kc * KC + c8);
            *reinterpret_cast<u32x4*>(&sB[cur][row * PITCH + c8]) =
                *reinterpret_cast<const u32x4*>(gB + (size_t)row * DDIM + kc * KC + c8);
        }
        __syncthreads();
#endif

        const u16* bufA = sA[cur];
        const u16* bufB = sB[cur];

        Frag a[2];
        #pragma unroll
        for (int tr = 0; tr < 2; ++tr) {
            const int base = (wm + tr * 16 + lr) * PITCH + half * 8;
            a[tr].u[0] = *reinterpret_cast<const u32x4*>(&bufA[base]);
            a[tr].u[1] = *reinterpret_cast<const u32x4*>(&bufA[base + 16]);
        }
        Frag b[4];
        #pragma unroll
        for (int tc = 0; tc < 4; ++tc) {
            const int base = (wn + tc * 16 + lr) * PITCH + half * 8;
            b[tc].u[0] = *reinterpret_cast<const u32x4*>(&bufB[base]);
            b[tc].u[1] = *reinterpret_cast<const u32x4*>(&bufB[base + 16]);
        }
        #pragma unroll
        for (int tr = 0; tr < 2; ++tr)
            #pragma unroll
            for (int tc = 0; tc < 4; ++tc)
                acc[tr][tc] = __builtin_amdgcn_wmma_f32_16x16x32_bf16(
                    false, a[tr].v, false, b[tc].v,
                    (short)0, acc[tr][tc], false, false);
    }

    // ---- epilogue: sum softplus(-(t*sim + bias)) over the tile, in log2 units
    const float b2 = -(*biasPtr) * LOG2E;
    float lsum = 0.f;

    if (blockIdx.x != blockIdx.y) {
        #pragma unroll
        for (int tr = 0; tr < 2; ++tr)
            #pragma unroll
            for (int tc = 0; tc < 4; ++tc)
                #pragma unroll
                for (int r = 0; r < 8; ++r)
                    lsum += softplus_log2(fmaf(LOG2E, acc[tr][tc][r], b2));
    } else {
        #pragma unroll
        for (int tr = 0; tr < 2; ++tr) {
            #pragma unroll
            for (int tc = 0; tc < 4; ++tc) {
                const int mBase = wm + tr * 16 + half * 8;
                const int nIdx  = wn + tc * 16 + lr;
                const int dd = nIdx - mBase;   // diagonal hits at r == dd
                #pragma unroll
                for (int r = 0; r < 8; ++r) {
                    const float sgn2 = (r == dd) ? -LOG2E : LOG2E;
                    lsum += softplus_log2(fmaf(sgn2, acc[tr][tc][r], b2));
                }
            }
        }
    }

    // wave32 reduce
    #pragma unroll
    for (int o = 16; o > 0; o >>= 1) lsum += __shfl_down(lsum, o, 32);
    if (lane == 0) sred[wid] = lsum;
    __syncthreads();
    if (t == 0) {
        float s = 0.f;
        #pragma unroll
        for (int i = 0; i < 8; ++i) s += sred[i];
        partials[blockIdx.y * gridDim.x + blockIdx.x] = s * LN2;
    }
}

// ---------------- final reduction ----------------
__global__ __launch_bounds__(256) void reduce_partials(const float* __restrict__ partials,
                                                       int n, float* __restrict__ out) {
    __shared__ float s[256];
    float v = 0.f;
    for (int i = threadIdx.x; i < n; i += 256) v += partials[i];
    s[threadIdx.x] = v;
    __syncthreads();
    for (int st = 128; st > 0; st >>= 1) {
        if (threadIdx.x < st) s[threadIdx.x] += s[threadIdx.x + st];
        __syncthreads();
    }
    if (threadIdx.x == 0) out[0] = s[0] / ((float)NTOT * (float)NTOT);
}

extern "C" void kernel_launch(void* const* d_in, const int* in_sizes, int n_in,
                              void* d_out, int out_size, void* d_ws, size_t ws_size,
                              hipStream_t stream) {
    const float* z_a  = (const float*)d_in[0];
    const float* z_b  = (const float*)d_in[1];
    const float* bias = (const float*)d_in[2];

    u16* Abf = (u16*)d_ws;
    u16* Bbf = Abf + (size_t)NTOT * DDIM;
    float* partials = (float*)((char*)d_ws + 2ull * NTOT * DDIM * sizeof(u16));

    const int nconv = (NTOT * DDIM) / 8;
    cvt_bf16<<<nconv / 256, 256, 0, stream>>>(z_a, z_b, Abf, Bbf);

    dim3 grid(NTOT / BN, NTOT / BM);                  // 128 x 128 blocks
    siglip_gemm_loss<<<grid, 256, 0, stream>>>(Abf, Bbf, bias, partials);

    reduce_partials<<<1, 256, 0, stream>>>(partials, (NTOT / BM) * (NTOT / BN),
                                           (float*)d_out);
}
